// SKIP_64922725646668
// MI455X (gfx1250) — matrix operand order repaired and checked by reading, not compiled, not run on hardware
//
#include <hip/hip_runtime.h>
#include <hip/hip_bf16.h>

// Problem constants (match reference)
#define NN      100000
#define EE      1600000
#define IN_DIM  92
#define HD      128
#define BG      512
#define GID     107
#define NL      5
#define BN_EPS  1e-5f

#define CDIV(a,b) (((a)+(b)-1)/(b))

typedef __attribute__((ext_vector_type(16))) __bf16 v16bf;
typedef __attribute__((ext_vector_type(8)))  float  v8f;

// ---------------------------------------------------------------------------
// Generic GEMM:  C[M,Nout] (fp32) = A[M,KTOT] (fp32) x Bt[Nout,KP] (bf16, B^T)
//                (+ bias[Nout]) (+ existing C if accum) (relu optional)
// One wave computes a 16 x (16*WT) tile via v_wmma_f32_16x16x32_bf16.
// KTOT is a template parameter -> K loop fully unrolled, tail guards
// constant-folded into clamped loads + selects (no branches, EXEC all-1s).
//
// Fragment layouts (wave32; A/B use this code's contiguous k order, same in both):
//  A (16x32, 16-bit): lane m=l&15, hi=l>>4;
//     halves 0..7  <- K = kk + 16*hi + t     (row-contiguous in A)
//     halves 8..15 <- K = kk + 16*hi + 8 + t
//  B (32x16, 16-bit): lane n=l&15 holds K = kk + 16*hi + (0..15),
//     contiguous in Bt row (Bt is B^T with KP-padded rows).
//  C/D: lane n=l&15; vgpr j holds row m0 + j + 8*hi.
// ---------------------------------------------------------------------------
template<int KTOT, int WT>
__global__ void __launch_bounds__(256)
k_gemm_bf16(const float* __restrict__ A, const __bf16* __restrict__ Bt,
            const float* __restrict__ bias, float* __restrict__ C,
            int M, int Nout, int lda, int ldc, int relu, int accum)
{
  constexpr int KP = (KTOT + 31) & ~31;
  const int lane = threadIdx.x & 31;
  const int wave = threadIdx.x >> 5;
  const int NG = Nout / (16 * WT);          // column groups per row strip
  const int MT = M >> 4;
  int grp = blockIdx.x * (blockDim.x >> 5) + wave;
  if (grp >= MT * NG) return;               // wave-uniform: EXEC stays all-1s
  const int rowT = grp / NG;
  const int colG = grp - rowT * NG;
  const int m0 = rowT << 4;
  const int n0 = colG * (16 * WT);
  const int ln = lane & 15;
  const int hi = lane >> 4;
  const int mrow = m0 + ln;

  const v8f vzero = {0.f,0.f,0.f,0.f,0.f,0.f,0.f,0.f};
  v8f acc[WT];
#pragma unroll
  for (int w = 0; w < WT; ++w) acc[w] = vzero;
  if (accum) {
#pragma unroll
    for (int w = 0; w < WT; ++w)
#pragma unroll
      for (int j = 0; j < 8; ++j)
        acc[w][j] = C[(size_t)(m0 + j + 8 * hi) * ldc + n0 + w * 16 + ln];
  }

  const float* arow = A + (size_t)mrow * lda;

#pragma unroll
  for (int kk = 0; kk < KP; kk += 32) {
    v16bf a;
    if (kk + 32 <= KTOT) {                  // constant-folded after unroll
#pragma unroll
      for (int t = 0; t < 8; ++t) {
        a[t]     = (__bf16)arow[kk + 16 * hi + t];
        a[8 + t] = (__bf16)arow[kk + 16 * hi + 8 + t];
      }
    } else {                                // ragged tail: clamped loads + select
#pragma unroll
      for (int t = 0; t < 8; ++t) {
        int c0 = kk + 16 * hi + t;
        int c1 = kk + 16 * hi + 8 + t;
        float v0 = arow[c0 < KTOT ? c0 : (KTOT - 1)];
        float v1 = arow[c1 < KTOT ? c1 : (KTOT - 1)];
        a[t]     = (c0 < KTOT) ? (__bf16)v0 : (__bf16)0.f;
        a[8 + t] = (c1 < KTOT) ? (__bf16)v1 : (__bf16)0.f;
      }
    }
#pragma unroll
    for (int w = 0; w < WT; ++w) {
      v16bf b = *reinterpret_cast<const v16bf*>(
          Bt + (size_t)(n0 + w * 16 + ln) * KP + kk + 16 * hi);
      acc[w] = __builtin_amdgcn_wmma_f32_16x16x32_bf16(
          false, a, false, b, (short)0, acc[w], false, false);
    }
  }

#pragma unroll
  for (int w = 0; w < WT; ++w) {
    float bv = bias ? bias[n0 + w * 16 + ln] : 0.f;
#pragma unroll
    for (int j = 0; j < 8; ++j) {
      float v = acc[w][j] + bv;
      if (relu) v = fmaxf(v, 0.f);
      C[(size_t)(m0 + j + 8 * hi) * ldc + n0 + w * 16 + ln] = v;
    }
  }
}

// ---------------------------------------------------------------------------
// Weight prep: fp32 -> bf16, optional transpose, K-pad with zeros.
// dst[n*Kp + k] = (k<Kin) ? (tr ? src[k*ld + n] : src[n*ld + k]) : 0
// ---------------------------------------------------------------------------
__global__ void k_prep_w(__bf16* dst, const float* src,
                         int Nout, int Kin, int Kp, int tr, int ld)
{
  int idx = blockIdx.x * blockDim.x + threadIdx.x;
  if (idx >= Nout * Kp) return;
  int n = idx / Kp, k = idx - n * Kp;
  float v = 0.f;
  if (k < Kin) v = tr ? src[(size_t)k * ld + n] : src[(size_t)n * ld + k];
  dst[idx] = (__bf16)v;
}

__global__ void k_vadd(float* o, const float* a, const float* b, int n)
{
  int i = blockIdx.x * blockDim.x + threadIdx.x;
  if (i < n) o[i] = a[i] + b[i];
}

// ---------------------------------------------------------------------------
// GCN normalization
// ---------------------------------------------------------------------------
__global__ void k_deg(const int* __restrict__ dst, const float* __restrict__ ed,
                      float* __restrict__ deg)
{
  int e = blockIdx.x * blockDim.x + threadIdx.x;
  if (e >= EE) return;
  atomicAdd(&deg[dst[e]], ed[e]);
}

__global__ void k_norm(float* __restrict__ deg_dinv, float* __restrict__ snorm)
{
  int i = blockIdx.x * blockDim.x + threadIdx.x;
  if (i >= NN) return;
  float di = rsqrtf(deg_dinv[i] + 2.0f);
  deg_dinv[i] = di;
  snorm[i] = 2.0f * di * di;
}

__global__ void k_enorm(const int* __restrict__ src, const int* __restrict__ dst,
                        const float* __restrict__ ed, const float* __restrict__ dinv,
                        float* __restrict__ en)
{
  int e = blockIdx.x * blockDim.x + threadIdx.x;
  if (e >= EE) return;
  en[e] = dinv[src[e]] * ed[e] * dinv[dst[e]];
}

__global__ void k_agg_init(float* __restrict__ agg, const float* __restrict__ xw,
                           const float* __restrict__ snorm, const float* __restrict__ cb)
{
  size_t idx = (size_t)blockIdx.x * blockDim.x + threadIdx.x;
  if (idx >= (size_t)NN * HD) return;
  int c = idx & (HD - 1);
  int n = idx >> 7;
  agg[idx] = snorm[n] * xw[idx] + cb[c];
}

// one wave per edge, 4 channels per lane (float4 gather + 4 atomics to L2)
__global__ void k_scatter(const float* __restrict__ xw, const int* __restrict__ src,
                          const int* __restrict__ dst, const float* __restrict__ en,
                          float* __restrict__ agg)
{
  int e = blockIdx.x * (blockDim.x >> 5) + (threadIdx.x >> 5);
  if (e >= EE) return;
  int lane = threadIdx.x & 31;
  int s = src[e], d = dst[e];
  float w = en[e];
  float4 v = reinterpret_cast<const float4*>(xw + (size_t)s * HD)[lane];
  float* ap = agg + (size_t)d * HD + lane * 4;
  atomicAdd(ap + 0, w * v.x);
  atomicAdd(ap + 1, w * v.y);
  atomicAdd(ap + 2, w * v.z);
  atomicAdd(ap + 3, w * v.w);
}

// ---------------------------------------------------------------------------
// BatchNorm (batch statistics over N rows, per channel)
// ---------------------------------------------------------------------------
__global__ void k_bn_stats(const float* __restrict__ agg, float* __restrict__ sums)
{
  int c = threadIdx.x;   // blockDim == 128
  float s = 0.f, s2 = 0.f;
  for (int r = blockIdx.x; r < NN; r += gridDim.x) {
    float v = agg[(size_t)r * HD + c];
    s += v; s2 += v * v;
  }
  atomicAdd(&sums[c], s);
  atomicAdd(&sums[HD + c], s2);
}

__global__ void k_bn_final(const float* __restrict__ sums, const float* __restrict__ g,
                           const float* __restrict__ b, float* __restrict__ ss)
{
  int c = threadIdx.x;   // 128 threads, 1 block
  float mean = sums[c] * (1.0f / NN);
  float var  = sums[HD + c] * (1.0f / NN) - mean * mean;
  float sc   = g[c] * rsqrtf(var + BN_EPS);
  ss[c]      = sc;
  ss[HD + c] = b[c] - mean * sc;
}

__global__ void k_bn_apply(float* __restrict__ x, const float* __restrict__ agg,
                           const float* __restrict__ ss)
{
  size_t idx = (size_t)blockIdx.x * blockDim.x + threadIdx.x;
  if (idx >= (size_t)NN * HD) return;
  int c = idx & (HD - 1);
  x[idx] = x[idx] + ss[c] * agg[idx] + ss[HD + c];   // BN + residual
}

// ---------------------------------------------------------------------------
// Attention scoring + segment softmax
// ---------------------------------------------------------------------------
__device__ inline unsigned fenc(float f) {
  unsigned u = __float_as_uint(f);
  return (u & 0x80000000u) ? ~u : (u | 0x80000000u);
}
__device__ inline float fdec(unsigned u) {
  return __uint_as_float((u & 0x80000000u) ? (u & 0x7fffffffu) : ~u);
}

__global__ void k_gat_e(const float* __restrict__ g, const float* __restrict__ W,
                        const float* __restrict__ b, float* __restrict__ e)
{
  int n = blockIdx.x * blockDim.x + threadIdx.x;
  if (n >= NN) return;
  const float* gr = g + (size_t)n * HD;
  float s = b[0];
#pragma unroll 8
  for (int c = 0; c < HD; ++c) s += gr[c] * W[c];
  e[n] = s;
}

__global__ void k_smax_max(const float* __restrict__ e, const int* __restrict__ seg,
                           unsigned* __restrict__ menc)
{
  int i = blockIdx.x * blockDim.x + threadIdx.x;
  if (i >= NN) return;
  atomicMax(&menc[seg[i]], fenc(e[i]));
}

__global__ void k_smax_sum(const float* __restrict__ e, const int* __restrict__ seg,
                           const unsigned* __restrict__ menc,
                           float* __restrict__ ex, float* __restrict__ S)
{
  int i = blockIdx.x * blockDim.x + threadIdx.x;
  if (i >= NN) return;
  int b = seg[i];
  float v = expf(e[i] - fdec(menc[b]));
  ex[i] = v;
  atomicAdd(&S[b], v);
}

__global__ void k_scale_x(float* __restrict__ x, const float* __restrict__ ex,
                          const float* __restrict__ S, const int* __restrict__ batch)
{
  size_t idx = (size_t)blockIdx.x * blockDim.x + threadIdx.x;
  if (idx >= (size_t)NN * HD) return;
  int n = idx >> 7;
  x[idx] *= ex[n] / S[batch[n]];
}

// ---------------------------------------------------------------------------
// Set2Set pieces
// ---------------------------------------------------------------------------
__device__ inline float sigm(float x) { return 1.0f / (1.0f + expf(-x)); }

__global__ void k_lstm(const float* __restrict__ G, float* __restrict__ hs,
                       float* __restrict__ cs)
{
  int idx = blockIdx.x * blockDim.x + threadIdx.x;
  if (idx >= BG * HD) return;
  int b = idx >> 7, h = idx & (HD - 1);
  const float* gr = G + (size_t)b * 4 * HD;
  float ig = sigm(gr[h]);
  float fg = sigm(gr[HD + h]);
  float gg = tanhf(gr[2 * HD + h]);
  float og = sigm(gr[3 * HD + h]);
  float c  = fg * cs[idx] + ig * gg;
  cs[idx] = c;
  hs[idx] = og * tanhf(c);
}

__global__ void k_en(const float* __restrict__ x, const float* __restrict__ hs,
                     const int* __restrict__ batch, float* __restrict__ e)
{
  int n = blockIdx.x * blockDim.x + threadIdx.x;
  if (n >= NN) return;
  const float* xr = x + (size_t)n * HD;
  const float* hr = hs + (size_t)batch[n] * HD;
  float s = 0.f;
#pragma unroll 8
  for (int c = 0; c < HD; ++c) s += xr[c] * hr[c];
  e[n] = s;
}

__global__ void k_raccum(const float* __restrict__ x, const float* __restrict__ ex,
                         const float* __restrict__ S, const int* __restrict__ batch,
                         float* __restrict__ r)
{
  size_t idx = (size_t)blockIdx.x * blockDim.x + threadIdx.x;
  if (idx >= (size_t)NN * HD) return;
  int n = idx >> 7, c = idx & (HD - 1);
  int b = batch[n];
  atomicAdd(&r[(size_t)b * HD + c], (ex[n] / S[b]) * x[idx]);
}

__global__ void k_qstar(const float* __restrict__ hs, const float* __restrict__ r,
                        float* __restrict__ q)
{
  int idx = blockIdx.x * blockDim.x + threadIdx.x;
  if (idx >= BG * HD) return;
  int b = idx >> 7, h = idx & (HD - 1);
  q[(size_t)b * 2 * HD + h]      = hs[idx];
  q[(size_t)b * 2 * HD + HD + h] = r[idx];
}

__global__ void k_out2(const float* __restrict__ y2, const float* __restrict__ W,
                       const float* __restrict__ b, float* __restrict__ out)
{
  int g = blockIdx.x * blockDim.x + threadIdx.x;
  if (g >= BG) return;
  float s = b[0];
#pragma unroll
  for (int c = 0; c < 32; ++c) s += y2[(size_t)g * 32 + c] * W[c];
  out[g] = s;
}

// ---------------------------------------------------------------------------
extern "C" void kernel_launch(void* const* d_in, const int* in_sizes, int n_in,
                              void* d_out, int out_size, void* d_ws, size_t ws_size,
                              hipStream_t stream)
{
  (void)in_sizes; (void)n_in; (void)out_size; (void)ws_size;

  const float* x_in    = (const float*)d_in[0];
  const int*   eidx    = (const int*)d_in[1];
  const int*   esrc    = eidx;
  const int*   edst    = eidx + EE;
  const float* edist   = (const float*)d_in[2];
  const float* ginfo   = (const float*)d_in[3];
  const int*   batch   = (const int*)d_in[4];
  const float* lin_W   = (const float*)d_in[5];
  const float* lin_b   = (const float*)d_in[6];
  const float* conv_W  = (const float*)d_in[7];
  const float* conv_b  = (const float*)d_in[8];
  const float* bn_g    = (const float*)d_in[9];
  const float* bn_b    = (const float*)d_in[10];
  const float* gatin_W = (const float*)d_in[11];
  const float* gatin_b = (const float*)d_in[12];
  const float* gatl_W  = (const float*)d_in[13];
  const float* gatl_b  = (const float*)d_in[14];
  const float* gato_W  = (const float*)d_in[15];
  const float* gato_b  = (const float*)d_in[16];
  const float* Wih     = (const float*)d_in[17];
  const float* Whh     = (const float*)d_in[18];
  const float* bih     = (const float*)d_in[19];
  const float* bhh     = (const float*)d_in[20];
  const float* mlp_W   = (const float*)d_in[21];
  const float* mlp_b   = (const float*)d_in[22];
  const float* o1_W    = (const float*)d_in[23];
  const float* o1_b    = (const float*)d_in[24];
  const float* o2_W    = (const float*)d_in[25];
  const float* o2_b    = (const float*)d_in[26];
  float* out = (float*)d_out;

  // workspace bump allocator (256B aligned)
  size_t off = 0;
  auto walloc = [&](size_t bytes) -> char* {
    off = (off + 255) & ~(size_t)255;
    char* p = (char*)d_ws + off;
    off += bytes;
    return p;
  };

  float* x_cur  = (float*)walloc(sizeof(float) * (size_t)NN * HD);
  float* xw     = (float*)walloc(sizeof(float) * (size_t)NN * HD);
  float* agg    = (float*)walloc(sizeof(float) * (size_t)NN * HD);
  float* dinv   = (float*)walloc(sizeof(float) * NN);      // deg -> dinv in place
  float* snorm  = (float*)walloc(sizeof(float) * NN);
  float* enorm  = (float*)walloc(sizeof(float) * EE);
  float* e_buf  = (float*)walloc(sizeof(float) * NN);
  float* ex_buf = (float*)walloc(sizeof(float) * NN);
  unsigned* menc = (unsigned*)walloc(sizeof(unsigned) * BG);
  float* Ssum   = (float*)walloc(sizeof(float) * BG);
  float* hs     = (float*)walloc(sizeof(float) * BG * HD);
  float* cs     = (float*)walloc(sizeof(float) * BG * HD);
  float* qstar  = (float*)walloc(sizeof(float) * BG * 2 * HD);
  float* gates  = (float*)walloc(sizeof(float) * BG * 4 * HD);
  float* rbuf   = (float*)walloc(sizeof(float) * BG * HD);
  float* y1     = (float*)walloc(sizeof(float) * BG * 2 * HD);
  float* y2     = (float*)walloc(sizeof(float) * BG * 32);
  float* bnsums = (float*)walloc(sizeof(float) * 2 * HD);
  float* bnss   = (float*)walloc(sizeof(float) * 2 * HD);
  float* lbias  = (float*)walloc(sizeof(float) * 4 * HD);

  __bf16* lin_Wt  = (__bf16*)walloc(2 * (size_t)HD * 96);
  __bf16* conv_Wt = (__bf16*)walloc(2 * (size_t)NL * HD * HD);
  __bf16* gatA    = (__bf16*)walloc(2 * (size_t)HD * HD);
  __bf16* gatB    = (__bf16*)walloc(2 * (size_t)HD * HD);
  __bf16* gatL0   = (__bf16*)walloc(2 * (size_t)HD * HD);
  __bf16* gatL1   = (__bf16*)walloc(2 * (size_t)HD * HD);
  __bf16* WihT    = (__bf16*)walloc(2 * (size_t)4 * HD * 2 * HD);
  __bf16* WhhT    = (__bf16*)walloc(2 * (size_t)4 * HD * HD);
  __bf16* mlpT    = (__bf16*)walloc(2 * (size_t)2 * HD * 2 * HD);
  __bf16* o1T     = (__bf16*)walloc(2 * (size_t)32 * 2 * HD);

  auto prep = [&](__bf16* dst, const float* src, int Nout, int Kin, int Kp, int tr, int ld) {
    int tot = Nout * Kp;
    k_prep_w<<<CDIV(tot, 256), 256, 0, stream>>>(dst, src, Nout, Kin, Kp, tr, ld);
  };

#define GEMM(KT, WT, A_, Bt_, bias_, C_, M_, Nout_, lda_, ldc_, relu_, accum_)          \
  k_gemm_bf16<KT, WT><<<CDIV(((M_) / 16) * ((Nout_) / (16 * (WT))), 8), 256, 0,         \
                        stream>>>(A_, Bt_, bias_, C_, M_, Nout_, lda_, ldc_, relu_, accum_)

  // ---- weight prep (bf16, transposed to Bt[n][k], K padded to 32) ----
  prep(lin_Wt, lin_W, HD, IN_DIM, 96, 1, HD);
  for (int l = 0; l < NL; ++l)
    prep(conv_Wt + (size_t)l * HD * HD, conv_W + (size_t)l * HD * HD, HD, HD, HD, 1, HD);
  prep(gatA, gatin_W, HD, HD, HD, 1, HD);
  prep(gatB, gatin_W + (size_t)HD * HD, HD, GID, HD, 1, HD);
  prep(gatL0, gatl_W, HD, HD, HD, 1, HD);
  prep(gatL1, gatl_W + (size_t)HD * HD, HD, HD, HD, 1, HD);
  prep(WihT, Wih, 4 * HD, 2 * HD, 2 * HD, 0, 2 * HD);
  prep(WhhT, Whh, 4 * HD, HD, HD, 0, HD);
  prep(mlpT, mlp_W, 2 * HD, 2 * HD, 2 * HD, 1, 2 * HD);
  prep(o1T, o1_W, 32, 2 * HD, 2 * HD, 1, 32);
  k_vadd<<<2, 256, 0, stream>>>(lbias, bih, bhh, 4 * HD);

  // ---- input projection: x = relu(x_in @ lin_W + b) ----
  GEMM(92, 4, x_in, lin_Wt, lin_b, x_cur, NN, HD, IN_DIM, HD, 1, 0);

  // ---- GCN normalization ----
  hipMemsetAsync(dinv, 0, sizeof(float) * NN, stream);
  k_deg<<<CDIV(EE, 256), 256, 0, stream>>>(edst, edist, dinv);
  k_norm<<<CDIV(NN, 256), 256, 0, stream>>>(dinv, snorm);
  k_enorm<<<CDIV(EE, 256), 256, 0, stream>>>(esrc, edst, edist, dinv, enorm);

  // ---- conv stack ----
  const int NE = (int)((size_t)NN * HD);
  for (int l = 0; l < NL; ++l) {
    GEMM(128, 4, x_cur, conv_Wt + (size_t)l * HD * HD, nullptr, xw, NN, HD, HD, HD, 0, 0);
    k_agg_init<<<CDIV(NE, 256), 256, 0, stream>>>(agg, xw, snorm, conv_b + l * HD);
    k_scatter<<<CDIV(EE, 8), 256, 0, stream>>>(xw, esrc, edst, enorm, agg);
    hipMemsetAsync(bnsums, 0, sizeof(float) * 2 * HD, stream);
    k_bn_stats<<<960, 128, 0, stream>>>(agg, bnsums);
    k_bn_final<<<1, 128, 0, stream>>>(bnsums, bn_g, bn_b, bnss);
    k_bn_apply<<<CDIV(NE, 256), 256, 0, stream>>>(x_cur, agg, bnss);
  }

  // ---- global attention (g buffers reuse xw/agg) ----
  float* g1 = xw;
  float* g2 = agg;
  GEMM(128, 4, x_cur, gatA, gatin_b, g1, NN, HD, HD, HD, 0, 0);
  GEMM(107, 4, ginfo, gatB, nullptr, g1, NN, HD, GID, HD, 1, 1);   // += gi@Wb, relu
  GEMM(128, 4, g1, gatL0, gatl_b, g2, NN, HD, HD, HD, 1, 0);
  GEMM(128, 4, g2, gatL1, gatl_b + HD, g1, NN, HD, HD, HD, 1, 0);
  k_gat_e<<<CDIV(NN, 256), 256, 0, stream>>>(g1, gato_W, gato_b, e_buf);
  hipMemsetAsync(menc, 0, sizeof(unsigned) * BG, stream);
  hipMemsetAsync(Ssum, 0, sizeof(float) * BG, stream);
  k_smax_max<<<CDIV(NN, 256), 256, 0, stream>>>(e_buf, batch, menc);
  k_smax_sum<<<CDIV(NN, 256), 256, 0, stream>>>(e_buf, batch, menc, ex_buf, Ssum);
  k_scale_x<<<CDIV(NE, 256), 256, 0, stream>>>(x_cur, ex_buf, Ssum, batch);

  // ---- Set2Set (3 steps) ----
  hipMemsetAsync(hs, 0, sizeof(float) * BG * HD, stream);
  hipMemsetAsync(cs, 0, sizeof(float) * BG * HD, stream);
  hipMemsetAsync(qstar, 0, sizeof(float) * BG * 2 * HD, stream);
  for (int t = 0; t < 3; ++t) {
    GEMM(256, 4, qstar, WihT, lbias, gates, BG, 4 * HD, 2 * HD, 4 * HD, 0, 0);
    GEMM(128, 4, hs, WhhT, nullptr, gates, BG, 4 * HD, HD, 4 * HD, 0, 1);
    k_lstm<<<CDIV(BG * HD, 256), 256, 0, stream>>>(gates, hs, cs);
    k_en<<<CDIV(NN, 256), 256, 0, stream>>>(x_cur, hs, batch, e_buf);
    hipMemsetAsync(menc, 0, sizeof(unsigned) * BG, stream);
    hipMemsetAsync(Ssum, 0, sizeof(float) * BG, stream);
    k_smax_max<<<CDIV(NN, 256), 256, 0, stream>>>(e_buf, batch, menc);
    k_smax_sum<<<CDIV(NN, 256), 256, 0, stream>>>(e_buf, batch, menc, ex_buf, Ssum);
    hipMemsetAsync(rbuf, 0, sizeof(float) * BG * HD, stream);
    k_raccum<<<CDIV(NE, 256), 256, 0, stream>>>(x_cur, ex_buf, Ssum, batch, rbuf);
    k_qstar<<<CDIV(BG * HD, 256), 256, 0, stream>>>(hs, rbuf, qstar);
  }

  // ---- MLP head ----
  GEMM(256, 4, qstar, mlpT, mlp_b, y1, BG, 2 * HD, 2 * HD, 2 * HD, 1, 0);
  GEMM(256, 2, y1, o1T, o1_b, y2, BG, 32, 2 * HD, 32, 1, 0);
  k_out2<<<CDIV(BG, 256), 256, 0, stream>>>(y2, o2_W, o2_b, out);

#undef GEMM
}
